// CAPSAttention_61830349193832
// MI455X (gfx1250) — matrix operand order repaired and checked by reading, not compiled, not run on hardware
//
#include <hip/hip_runtime.h>
#include <hip/hip_bf16.h>

// ---------------- problem constants ----------------
constexpr int cB = 2, cT = 2048, cD = 2048, cH = 16, cDH = 128, cDC = 384;
constexpr int cMT = cB * cT; // 4096 rows

// ---------------- WMMA vector types ----------------
typedef __bf16 bf16_t;
typedef bf16_t  v16bf __attribute__((ext_vector_type(16)));
typedef float   v8f   __attribute__((ext_vector_type(8)));
typedef unsigned int v8u __attribute__((ext_vector_type(8)));
typedef unsigned int tdm_v4u __attribute__((ext_vector_type(4)));
typedef int          tdm_v8i __attribute__((ext_vector_type(8)));
typedef int          tdm_v4i __attribute__((ext_vector_type(4)));

__device__ __forceinline__ unsigned short f2bf(float f) {
    unsigned u = __builtin_bit_cast(unsigned, f);
    unsigned r = u + 0x7fffu + ((u >> 16) & 1u);   // round-to-nearest-even
    return (unsigned short)(r >> 16);
}
__device__ __forceinline__ unsigned pack2bf(float lo, float hi) {
    return (unsigned)f2bf(lo) | ((unsigned)f2bf(hi) << 16);
}
__device__ __forceinline__ v8f wmma_bf16(v8u a, v8u b, v8f c) {
    return __builtin_amdgcn_wmma_f32_16x16x32_bf16(
        false, __builtin_bit_cast(v16bf, a),
        false, __builtin_bit_cast(v16bf, b),
        (short)0, c, false, false);
}
__device__ __forceinline__ float softplus_f(float x) {
    return x > 20.f ? x : log1pf(__expf(x));
}
__device__ __forceinline__ v8u make_v8u(uint4 a, uint4 b) {
    v8u r; r[0]=a.x; r[1]=a.y; r[2]=a.z; r[3]=a.w; r[4]=b.x; r[5]=b.y; r[6]=b.z; r[7]=b.w;
    return r;
}

// ---------------- Tensor Data Mover: 2D bf16 tile -> LDS ----------------
__device__ __forceinline__ void tdm_load_tile(const unsigned short* src,
                                              unsigned lds_off, int nrows,
                                              int Ntot, int K) {
    unsigned long long ga = (unsigned long long)(uintptr_t)src;
    tdm_v4u g0;
    g0[0] = 1u;                                        // count=1, user mode
    g0[1] = lds_off;                                   // lds_addr (bytes)
    g0[2] = (unsigned)(ga & 0xffffffffu);              // global_addr lo
    g0[3] = (unsigned)((ga >> 32) & 0x01ffffffu) | (2u << 30);  // addr hi | type=2
    unsigned dim0 = (unsigned)K, dim1 = (unsigned)Ntot;
    unsigned tile0 = 32u, tile1 = (unsigned)nrows;
    unsigned long long st0 = (unsigned long long)K;    // tensor_dim0_stride
    tdm_v8i g1;
    g1[0] = (int)0x00010000u;                          // data_size = 2 bytes
    g1[1] = (int)((dim0 & 0xffffu) << 16);
    g1[2] = (int)(((dim0 >> 16) & 0xffffu) | ((dim1 & 0xffffu) << 16));
    g1[3] = (int)(((dim1 >> 16) & 0xffffu) | (tile0 << 16));
    g1[4] = (int)tile1;
    g1[5] = (int)(unsigned)(st0 & 0xffffffffu);
    g1[6] = (int)(unsigned)((st0 >> 32) & 0xffffu);
    g1[7] = 0;
    tdm_v4i z = {0, 0, 0, 0};
#if __clang_major__ >= 23
    tdm_v8i z8 = {0, 0, 0, 0, 0, 0, 0, 0};
    __builtin_amdgcn_tensor_load_to_lds(g0, g1, z, z, z8, 0);
#else
    __builtin_amdgcn_tensor_load_to_lds(g0, g1, z, z, 0);
#endif
}

// =====================================================================
// Kernel 0a: elementwise f32 -> bf16 (n multiple of 4)
// =====================================================================
__global__ __launch_bounds__(256) void cvt_bf16(const float* __restrict__ src,
                                                unsigned short* __restrict__ dst,
                                                long n) {
    long i = ((long)blockIdx.x * blockDim.x + threadIdx.x) * 4;
    if (i < n) {
        float4 f = *(const float4*)(src + i);
        uint2 o; o.x = pack2bf(f.x, f.y); o.y = pack2bf(f.z, f.w);
        *(uint2*)(dst + i) = o;
    }
}

// =====================================================================
// Kernel 0b: W (KxN f32) -> Wt (NxK bf16) transpose via LDS tile
// =====================================================================
__global__ __launch_bounds__(256) void transpose_bf16(const float* __restrict__ W,
                                                      unsigned short* __restrict__ Wt,
                                                      int K, int N) {
    __shared__ float tile[32][33];
    const int k0 = blockIdx.x * 32, n0 = blockIdx.y * 32;
    for (int r = threadIdx.y; r < 32; r += 8) {
        int k = k0 + r, n = n0 + threadIdx.x;
        tile[r][threadIdx.x] = (k < K && n < N) ? W[(size_t)k * N + n] : 0.f;
    }
    __syncthreads();
    for (int r = threadIdx.y; r < 32; r += 8) {
        int n = n0 + r, k = k0 + threadIdx.x;
        if (n < N && k < K) Wt[(size_t)n * K + k] = f2bf(tile[threadIdx.x][r]);
    }
}

// =====================================================================
// Kernel 1: Y(MxN,f32) = Xb(MxK,bf16) @ Wt(NxK,bf16)^T
// TDM-staged, double-buffered, peeled 64-wide K loop.
// =====================================================================
template<int NT>
__global__ __launch_bounds__(256) void gemm_tn(
    const unsigned short* __restrict__ Xb, const unsigned short* __restrict__ Wt,
    float* __restrict__ Y, int M, int N, int K)
{
    __shared__ unsigned short ldsW[2][128 * 32];       // 2 x 8 KB at LDS offsets 0/8192
    const int lane = threadIdx.x, w = threadIdx.y;
    const int m = lane & 15, hi = lane >> 4;
    const int nBlocks = (N + 127) >> 7;
    const int bM = blockIdx.x / nBlocks;
    const int n0 = (blockIdx.x % nBlocks) * 128;
    const unsigned short* xrow = Xb + (size_t)(bM * 128 + w * 16 + m) * K + 16 * hi;
    const unsigned short* wsrc = Wt + (size_t)n0 * K;

    v8f acc[NT] = {};
    if (w == 0) {                                      // prologue: fill both buffers
        tdm_load_tile(wsrc,      0u,    NT * 16, N, K);
        tdm_load_tile(wsrc + 32, 8192u, NT * 16, N, K);
    }

    auto phase = [&](int kk, int buf, int wcnt, bool issueNext) {
        if (w == 0) {
            if (wcnt) __builtin_amdgcn_s_wait_tensorcnt((short)1);
            else      __builtin_amdgcn_s_wait_tensorcnt((short)0);
        }
        __syncthreads();                               // tile visible to all waves
        const uint4* ap = (const uint4*)(xrow + kk);
        v8u au = make_v8u(ap[0], ap[1]);               // A frag: two b128 loads
#pragma unroll
        for (int j = 0; j < NT; ++j) {
            const uint4* bp = (const uint4*)&ldsW[buf][(j * 16 + m) * 32 + 16 * hi];
            acc[j] = wmma_bf16(au, make_v8u(bp[0], bp[1]), acc[j]);
        }
        __syncthreads();                               // all reads done before refill
        if (issueNext && w == 0)
            tdm_load_tile(wsrc + kk + 64, buf ? 8192u : 0u, NT * 16, N, K);
    };

    int kk = 0;
    for (; kk + 64 < K; kk += 64) {                    // steady state: 2 tiles in flight
        phase(kk,      0, 1, true);
        phase(kk + 32, 1, 1, true);
    }
    phase(kk,      0, 1, false);                       // tail: exact waits
    phase(kk + 32, 1, 0, false);

#pragma unroll
    for (int j = 0; j < NT; ++j) {
        const int n = n0 + j * 16 + m;
#pragma unroll
        for (int r = 0; r < 8; ++r)
            Y[(size_t)(bM * 128 + w * 16 + r + 8 * hi) * N + n] = acc[j][r];
    }
}

// =====================================================================
// Kernel 2: per-(b,h) scan: p_max pass, cumsums + RoPE; emits
//   q_cat/k_cat bf16 (B,H,T,384) and Vt bf16 (B,H,128,T).
// =====================================================================
__global__ __launch_bounds__(128) void scan_kernel(
    const float* __restrict__ qf, const float* __restrict__ kf,
    const float* __restrict__ vf, const float* __restrict__ pf,
    const float* __restrict__ gf, const float* __restrict__ cf,
    unsigned short* __restrict__ Qc, unsigned short* __restrict__ Kc,
    unsigned short* __restrict__ Vt)
{
    const int bh = blockIdx.x;
    const int b = bh / cH, h = bh % cH;
    const int d = threadIdx.x;
    const float invfreq = powf(10000.f, -(float)(2 * (d & 63)) / (float)cDH);

    float pmax = -3.0e38f;
    for (int t = 0; t < cT; ++t) {
        const size_t row = (size_t)b * cT + t;
        const float cl = softplus_f(cf[row * cH + h]) + 1e-6f;
        pmax = fmaxf(pmax, pf[row * cD + h * cDH + d] + __logf(cl));
    }

    float p_cs = 0.f, g_cum = 0.f, clock_cs = 0.f;
    for (int t = 0; t < cT; ++t) {
        const size_t row = (size_t)b * cT + t;
        const size_t idx = row * cD + h * cDH + d;
        const float cl = softplus_f(cf[row * cH + h]) + 1e-6f;

        const float q = qf[idx], k = kf[idx], v = vf[idx];
        float sn, cs;
        __sincosf((float)t * invfreq, &sn, &cs);
        const float qp = __shfl_xor(q, 1);
        const float kp = __shfl_xor(k, 1);
        const float qr = (d & 1) ? fmaf(qp, sn, q * cs) : fmaf(-qp, sn, q * cs);
        const float kr = (d & 1) ? fmaf(kp, sn, k * cs) : fmaf(-kp, sn, k * cs);

        const float pexp = __expf(pf[idx] + __logf(cl) - pmax);
        p_cs += pexp;
        g_cum += -softplus_f(gf[idx]) * cl;
        const float gcp = __expf(fminf(fmaxf(g_cum, -50.f), 40.f));
        clock_cs += cl;

        const size_t o = ((size_t)bh * cT + t) * cDC + d;
        Qc[o]       = f2bf(qr / (p_cs + 1e-8f));
        Qc[o + 128] = f2bf(qr * gcp);
        Qc[o + 256] = f2bf(qr / clock_cs);
        Kc[o]       = f2bf(kr * pexp);
        Kc[o + 128] = f2bf(kr / (gcp + 1e-8f));
        Kc[o + 256] = f2bf(kr * cl);
        Vt[((size_t)bh * cDH + d) * cT + t] = f2bf(v);
    }
}

// =====================================================================
// Kernel 3: attention with online softmax, software-pipelined fragment
// loads so global b128 loads overlap WMMA issue.
// =====================================================================
__global__ __launch_bounds__(256) void attn_kernel(
    const unsigned short* __restrict__ Qc, const unsigned short* __restrict__ Kc,
    const unsigned short* __restrict__ Vt, unsigned short* __restrict__ Ob)
{
    __shared__ unsigned short pT[8][16 * 40];          // per-wave 16x32 P tile, stride 40
    const int lane = threadIdx.x, w = threadIdx.y;
    const int m = lane & 15, hi = lane >> 4;
    const int rowTiles = cT / 16;
    const int task = blockIdx.x * 8 + w;
    const int bh = task / rowTiles, tq = task % rowTiles;
    const int b = bh >> 4, h = bh & 15;

    const unsigned short* Qb = Qc + (size_t)bh * cT * cDC;
    const unsigned short* Kb = Kc + (size_t)bh * cT * cDC;
    const unsigned short* Vb = Vt + (size_t)bh * cDH * cT;

    v8u qfr[12];
#pragma unroll
    for (int kc = 0; kc < 12; ++kc) {
        const uint4* qp = (const uint4*)(Qb + (size_t)(tq * 16 + m) * cDC + kc * 32 + 16 * hi);
        qfr[kc] = make_v8u(qp[0], qp[1]);
    }

    float mr[8], lr[8];
    v8f oa[8] = {};
#pragma unroll
    for (int r = 0; r < 8; ++r) { mr[r] = -3.0e38f; lr[r] = 0.f; }

    const float scale = 0.05103103630798288f;          // 1/sqrt(384)

    for (int s0 = 0; s0 < cT; s0 += 32) {
        if (s0 + 32 < cT) {                            // gfx1250 global_prefetch_b8
            __builtin_prefetch(Kb + (size_t)(s0 + 32 + m) * cDC, 0, 1);
            __builtin_prefetch(Vb + (size_t)m * cT + s0 + 32, 0, 1);
        }
        const unsigned short* krow0 = Kb + (size_t)(s0 + m) * cDC + 16 * hi;
        const unsigned short* krow1 = Kb + (size_t)(s0 + 16 + m) * cDC + 16 * hi;

        // ---- QK^T: software-pipelined B fragments ----
        v8f c0 = {}, c1 = {};
        v8u bcur0, bcur1, bnxt0, bnxt1;
        {
            const uint4* p0 = (const uint4*)krow0;
            const uint4* p1 = (const uint4*)krow1;
            bcur0 = make_v8u(p0[0], p0[1]);
            bcur1 = make_v8u(p1[0], p1[1]);
        }
#pragma unroll
        for (int kc = 0; kc < 12; ++kc) {
            if (kc + 1 < 12) {                         // issue next loads before MMA
                const uint4* p0 = (const uint4*)(krow0 + (kc + 1) * 32);
                const uint4* p1 = (const uint4*)(krow1 + (kc + 1) * 32);
                bnxt0 = make_v8u(p0[0], p0[1]);
                bnxt1 = make_v8u(p1[0], p1[1]);
            }
            c0 = wmma_bf16(qfr[kc], bcur0, c0);
            c1 = wmma_bf16(qfr[kc], bcur1, c1);
            bcur0 = bnxt0; bcur1 = bnxt1;
        }

        // ---- online softmax (row = r + 8*hi, col = lane&15) ----
#pragma unroll
        for (int r = 0; r < 8; ++r) {
            const float x0 = c0[r] * scale, x1 = c1[r] * scale;
            float rm = fmaxf(x0, x1);
            rm = fmaxf(rm, __shfl_xor(rm, 1));
            rm = fmaxf(rm, __shfl_xor(rm, 2));
            rm = fmaxf(rm, __shfl_xor(rm, 4));
            rm = fmaxf(rm, __shfl_xor(rm, 8));
            const float mn  = fmaxf(mr[r], rm);
            const float esc = __expf(mr[r] - mn);
            const float p0 = __expf(x0 - mn), p1 = __expf(x1 - mn);
            float rs = p0 + p1;
            rs += __shfl_xor(rs, 1);
            rs += __shfl_xor(rs, 2);
            rs += __shfl_xor(rs, 4);
            rs += __shfl_xor(rs, 8);
            lr[r] = lr[r] * esc + rs;
            mr[r] = mn;
#pragma unroll
            for (int j = 0; j < 8; ++j) oa[j][r] *= esc;
            const int row = r + 8 * hi;
            pT[w][row * 40 + m]      = f2bf(p0);
            pT[w][row * 40 + 16 + m] = f2bf(p1);
        }
        asm volatile("s_wait_dscnt 0x0" ::: "memory");
        const uint4* pp = (const uint4*)&pT[w][m * 40 + 16 * hi];
        v8u pa = make_v8u(pp[0], pp[1]);

        // ---- O += P @ V: software-pipelined V fragments ----
        v8u vcur, vnxt;
        {
            const uint4* vp = (const uint4*)(Vb + (size_t)m * cT + s0 + 16 * hi);
            vcur = make_v8u(vp[0], vp[1]);
        }
#pragma unroll
        for (int j = 0; j < 8; ++j) {
            if (j + 1 < 8) {
                const uint4* vp = (const uint4*)(Vb + (size_t)((j + 1) * 16 + m) * cT + s0 + 16 * hi);
                vnxt = make_v8u(vp[0], vp[1]);
            }
            oa[j] = wmma_bf16(pa, vcur, oa[j]);
            vcur = vnxt;
        }
    }

#pragma unroll
    for (int j = 0; j < 8; ++j)
#pragma unroll
        for (int r = 0; r < 8; ++r) {
            const int trow = tq * 16 + r + 8 * hi;
            Ob[((size_t)(b * cT + trow)) * cD + h * cDH + j * 16 + m] =
                f2bf(oa[j][r] / lr[r]);
        }
}

// =====================================================================
extern "C" void kernel_launch(void* const* d_in, const int* in_sizes, int n_in,
                              void* d_out, int out_size, void* d_ws, size_t ws_size,
                              hipStream_t stream) {
    const float* x   = (const float*)d_in[0];
    const float* Wq  = (const float*)d_in[1];
    const float* Wk  = (const float*)d_in[2];
    const float* Wv  = (const float*)d_in[3];
    const float* Wg  = (const float*)d_in[4];
    const float* Wp  = (const float*)d_in[5];
    const float* Wcl = (const float*)d_in[6];
    const float* Wc  = (const float*)d_in[7];
    float* out = (float*)d_out;

    char* ws = (char*)d_ws;
    const size_t szMD  = (size_t)cMT * cD * sizeof(float);
    const size_t szMDb = (size_t)cMT * cD * 2;
    const size_t szWb  = (size_t)cD * cD * 2;
    const size_t szC   = ((size_t)cMT * cH * sizeof(float) + 255) & ~(size_t)255;
    const size_t szQC  = (size_t)cB * cH * cT * cDC * 2;
    const size_t szVT  = (size_t)cB * cH * cDH * cT * 2;
    size_t off = 0;
    unsigned short* Xb  = (unsigned short*)(ws + off); off += szMDb;
    unsigned short* Wt[6];
    for (int i = 0; i < 6; ++i) { Wt[i] = (unsigned short*)(ws + off); off += szWb; }
    unsigned short* Wtcl = (unsigned short*)(ws + off); off += ((size_t)cH * cD * 2 + 255) & ~(size_t)255;
    float* qf = (float*)(ws + off); off += szMD;
    float* kf = (float*)(ws + off); off += szMD;
    float* vf = (float*)(ws + off); off += szMD;
    float* pf = (float*)(ws + off); off += szMD;
    float* gf = (float*)(ws + off); off += szMD;
    float* cf = (float*)(ws + off); off += szC;
    unsigned short* Qc = (unsigned short*)(ws + off); off += szQC;
    unsigned short* Kc = (unsigned short*)(ws + off); off += szQC;
    unsigned short* Vt = (unsigned short*)(ws + off); off += szVT;
    unsigned short* attnb = (unsigned short*)qf;       // reuse: qf dead after scan

    dim3 blk(32, 8);

    // ---- precision conversion / weight transposes ----
    {
        long n = (long)cMT * cD;
        cvt_bf16<<<dim3((unsigned)((n / 4 + 255) / 256)), dim3(256), 0, stream>>>(x, Xb, n);
    }
    dim3 tgrid(cD / 32, cD / 32);
    transpose_bf16<<<tgrid, blk, 0, stream>>>(Wq, Wt[0], cD, cD);
    transpose_bf16<<<tgrid, blk, 0, stream>>>(Wk, Wt[1], cD, cD);
    transpose_bf16<<<tgrid, blk, 0, stream>>>(Wv, Wt[2], cD, cD);
    transpose_bf16<<<tgrid, blk, 0, stream>>>(Wp, Wt[3], cD, cD);
    transpose_bf16<<<tgrid, blk, 0, stream>>>(Wg, Wt[4], cD, cD);
    transpose_bf16<<<tgrid, blk, 0, stream>>>(Wc, Wt[5], cD, cD);
    transpose_bf16<<<dim3(cD / 32, 1), blk, 0, stream>>>(Wcl, Wtcl, cD, cH);

    // ---- projections ----
    const dim3 ggrid((cMT / 128) * (cD / 128));        // 512 blocks
    gemm_tn<8><<<ggrid, blk, 0, stream>>>(Xb, Wt[0], qf, cMT, cD, cD);
    gemm_tn<8><<<ggrid, blk, 0, stream>>>(Xb, Wt[1], kf, cMT, cD, cD);
    gemm_tn<8><<<ggrid, blk, 0, stream>>>(Xb, Wt[2], vf, cMT, cD, cD);
    gemm_tn<8><<<ggrid, blk, 0, stream>>>(Xb, Wt[3], pf, cMT, cD, cD);
    gemm_tn<8><<<ggrid, blk, 0, stream>>>(Xb, Wt[4], gf, cMT, cD, cD);
    gemm_tn<1><<<dim3(cMT / 128), blk, 0, stream>>>(Xb, Wtcl, cf, cMT, cH, cD);

    // ---- per-(b,h) scans + RoPE + bf16 packing ----
    scan_kernel<<<dim3(cB * cH), dim3(128), 0, stream>>>(qf, kf, vf, pf, gf, cf, Qc, Kc, Vt);

    // ---- attention (bf16 output into attnb == qf) ----
    const int tasks = cB * cH * (cT / 16);
    attn_kernel<<<dim3(tasks / 8), blk, 0, stream>>>(Qc, Kc, Vt, attnb);

    // ---- output projection ----
    gemm_tn<8><<<ggrid, blk, 0, stream>>>(attnb, Wt[5], out, cMT, cD, cD);
}
